// VehicleLSTM_29755533426778
// MI455X (gfx1250) — compile-verified, matrix operand order checked by reading, not run on hardware
//
#include <hip/hip_runtime.h>
#include <hip/hip_bf16.h>

// ---- problem constants (match reference) ----
#define B_ 256
#define S_ 512
#define IN_ 64
#define H_ 1024
#define G4H_ 4096   // 4*H
#define HEAD_ 128
#define OUT_ 64

typedef __attribute__((ext_vector_type(16))) __bf16 v16bf;
typedef __attribute__((ext_vector_type(8)))  float  v8f;

// ---------- helpers ----------
__device__ __forceinline__ __bf16 f2bf(float f) {
    unsigned u = __builtin_bit_cast(unsigned, f);
    unsigned r = u + 0x7FFFu + ((u >> 16) & 1u);   // round-to-nearest-even
    unsigned short h = (unsigned short)(r >> 16);
    return __builtin_bit_cast(__bf16, h);
}

__device__ __forceinline__ float sigmoidf(float x) {
    return 1.0f / (1.0f + __expf(-x));
}
__device__ __forceinline__ float tanh_fast(float x) {
    x = fminf(fmaxf(x, -15.0f), 15.0f);
    float e = __expf(2.0f * x);
    return (e - 1.0f) / (e + 1.0f);
}

// Load one 16x32 (A, MxK) or [16 N-rows x 32 K] (B) bf16 fragment.
// contiguous-K order chosen here for both A and B (k = 16*(lane>>4) + s): lane&15 selects the row (M or N),
// lane>>4 selects the 16-element K half; 32 contiguous bytes per lane.
__device__ __forceinline__ v16bf ldfrag(const __bf16* __restrict__ p, int stride) {
    int lane = threadIdx.x & 31;
    const __bf16* q = p + (size_t)(lane & 15) * stride + ((lane >> 4) << 4);
    return *reinterpret_cast<const v16bf*>(q);
}

__device__ __forceinline__ v8f wmma_bf16(v16bf a, v16bf b, v8f c) {
    return __builtin_amdgcn_wmma_f32_16x16x32_bf16(
        /*neg_a=*/false, a, /*neg_b=*/false, b,
        /*c_mod=*/(short)0, c, /*reuse_a=*/false, /*reuse_b=*/false);
}

// ---------- pre-processing kernels ----------
__global__ void cvt_bf16_kernel(const float* __restrict__ src, __bf16* __restrict__ dst, int n) {
    int i = blockIdx.x * 256 + threadIdx.x;
    if (i < n) dst[i] = f2bf(src[i]);
}

__global__ void bias_sum_kernel(const float* __restrict__ a, const float* __restrict__ b,
                                float* __restrict__ o, int n) {
    int i = blockIdx.x * 256 + threadIdx.x;
    if (i < n) o[i] = a[i] + b[i];
}

__global__ void state_zero_kernel(__bf16* __restrict__ h, float* __restrict__ c, int n) {
    int i = blockIdx.x * 256 + threadIdx.x;
    if (i < n) { h[i] = f2bf(0.0f); c[i] = 0.0f; }
}

// ---------- per-step LSTM cell kernel ----------
// grid = (H/64, B/32), block = 256 threads (8 waves).
// Block (jx, by) computes gates[32 batch x (4 gates x 64 hidden cols)] + cell update.
// Wave w (0..7): gate = w>>1, covers 32 gate columns ((w&1)*32..+32) x 32 batch rows
// as a 2x2 grid of 16x16 WMMA tiles (4 accumulators): 2 A-frags + 2 B-frags -> 4 WMMAs.
__global__ __launch_bounds__(256)
void lstm_step_kernel(const __bf16* __restrict__ xbf,   // [B,S,IN]
                      const __bf16* __restrict__ Wih,   // [4H,IN]
                      const __bf16* __restrict__ Whh,   // [4H,H]
                      const float*  __restrict__ bsum,  // [4H] = b_ih + b_hh
                      const __bf16* __restrict__ h_in,  // [B,H]
                      __bf16*       __restrict__ h_out, // [B,H]
                      float*        __restrict__ c,     // [B,H]
                      int s)
{
    const int tid  = threadIdx.x;
    const int lane = tid & 31;
    const int w    = tid >> 5;
    const int j0   = blockIdx.x * 64;   // hidden column base
    const int bm0  = blockIdx.y * 32;   // batch row base (32-row tile)
    const int gate = w >> 1;
    const int gc0  = gate * H_ + j0 + (w & 1) * 32;  // first of 32 gate-output rows in W

    v8f acc00 = {};  // M rows [bm0..+16),  N cols [gc0..+16)
    v8f acc01 = {};  // M rows [bm0..+16),  N cols [gc0+16..+32)
    v8f acc10 = {};  // M rows [bm0+16..),  N cols [gc0..+16)
    v8f acc11 = {};  // M rows [bm0+16..),  N cols [gc0+16..+32)

    // input projection xg = x_t @ W_ih^T   (K over IN=64)
    {
        const __bf16* xrow0 = xbf + (size_t)bm0 * (S_ * IN_) + (size_t)s * IN_;
        const __bf16* xrow1 = xrow0 + (size_t)16 * (S_ * IN_);
#pragma unroll
        for (int k0 = 0; k0 < IN_; k0 += 32) {
            v16bf a0 = ldfrag(xrow0 + k0, S_ * IN_);
            v16bf a1 = ldfrag(xrow1 + k0, S_ * IN_);
            v16bf b0 = ldfrag(Wih + (size_t)gc0 * IN_ + k0, IN_);
            v16bf b1 = ldfrag(Wih + (size_t)(gc0 + 16) * IN_ + k0, IN_);
            acc00 = wmma_bf16(a0, b0, acc00);
            acc01 = wmma_bf16(a0, b1, acc01);
            acc10 = wmma_bf16(a1, b0, acc10);
            acc11 = wmma_bf16(a1, b1, acc11);
        }
    }
    // recurrent projection h_prev @ W_hh^T   (K over H=1024)
    {
        const __bf16* hrow0 = h_in + (size_t)bm0 * H_;
        const __bf16* hrow1 = hrow0 + (size_t)16 * H_;
#pragma unroll 2
        for (int k0 = 0; k0 < H_; k0 += 32) {
            v16bf a0 = ldfrag(hrow0 + k0, H_);
            v16bf a1 = ldfrag(hrow1 + k0, H_);
            v16bf b0 = ldfrag(Whh + (size_t)gc0 * H_ + k0, H_);
            v16bf b1 = ldfrag(Whh + (size_t)(gc0 + 16) * H_ + k0, H_);
            acc00 = wmma_bf16(a0, b0, acc00);
            acc01 = wmma_bf16(a0, b1, acc01);
            acc10 = wmma_bf16(a1, b0, acc10);
            acc11 = wmma_bf16(a1, b1, acc11);
        }
    }

    // gate exchange through LDS: gl[m][gate*64 + jo], m = batch row within 32-row tile
    __shared__ __align__(32) float gl[32 * 256];
    {
        const int n  = lane & 15;
        const int mh = (lane >> 4) * 8;          // C layout: M = vgpr + 8*(lane>=16)
        const int cb = w * 32;
        const float biasA = bsum[gc0 + n];
        const float biasB = bsum[gc0 + 16 + n];
#pragma unroll
        for (int r = 0; r < 8; ++r) {
            const int m0 = mh + r;
            gl[m0 * 256 + cb + n]             = acc00[r] + biasA;
            gl[m0 * 256 + cb + 16 + n]        = acc01[r] + biasB;
            gl[(16 + m0) * 256 + cb + n]      = acc10[r] + biasA;
            gl[(16 + m0) * 256 + cb + 16 + n] = acc11[r] + biasB;
        }
    }
    __syncthreads();

    // LSTM cell: each thread handles 8 (batch,hidden) elements of the 32x64 tile
#pragma unroll
    for (int idx = tid; idx < 32 * 64; idx += 256) {
        const int m  = idx >> 6;
        const int jo = idx & 63;
        const float gi = gl[m * 256 + jo];         // i
        const float gf = gl[m * 256 + 64 + jo];    // f
        const float gg = gl[m * 256 + 128 + jo];   // g
        const float go = gl[m * 256 + 192 + jo];   // o
        const size_t gidx = (size_t)(bm0 + m) * H_ + j0 + jo;
        const float cold = c[gidx];
        const float cn = sigmoidf(gf) * cold + sigmoidf(gi) * tanh_fast(gg);
        const float hn = sigmoidf(go) * tanh_fast(cn);
        c[gidx]     = cn;
        h_out[gidx] = f2bf(hn);
    }
}

// ---------- per-step dense head ----------
// grid = B/16 blocks, 256 threads (8 waves).
// GEMM1: t[16x128] = relu(h @ W_dense^T + b_dense); GEMM2: out[16x64] = t @ W_dense2^T + b_dense2
__global__ __launch_bounds__(256)
void head_kernel(const __bf16* __restrict__ h,     // [B,H]
                 const __bf16* __restrict__ Wd,    // [HEAD,H]
                 const float*  __restrict__ bd,    // [HEAD]
                 const __bf16* __restrict__ Wd2,   // [OUT,HEAD]
                 const float*  __restrict__ bd2,   // [OUT]
                 float*        __restrict__ out,   // [B,S,OUT]
                 int s)
{
    const int tid  = threadIdx.x;
    const int lane = tid & 31;
    const int w    = tid >> 5;
    const int bm0  = blockIdx.x * 16;

    __shared__ __align__(32) __bf16 tl[16 * 128];

    // GEMM1: wave w computes cols [w*16, w*16+16)
    v8f acc = {};
    const __bf16* hrow = h + (size_t)bm0 * H_;
#pragma unroll 4
    for (int k0 = 0; k0 < H_; k0 += 32) {
        v16bf a = ldfrag(hrow + k0, H_);
        v16bf b = ldfrag(Wd + (size_t)(w * 16) * H_ + k0, H_);
        acc = wmma_bf16(a, b, acc);
    }
    const int n  = lane & 15;
    const int mh = (lane >> 4) * 8;
    {
        const float bias = bd[w * 16 + n];
#pragma unroll
        for (int r = 0; r < 8; ++r) {
            float v = acc[r] + bias;
            v = v > 0.0f ? v : 0.0f;               // relu
            tl[(mh + r) * 128 + w * 16 + n] = f2bf(v);
        }
    }
    __syncthreads();

    // GEMM2: waves 0..3 compute out cols [w*16, w*16+16) over K=128
    if (w < 4) {
        v8f acc2 = {};
#pragma unroll
        for (int k0 = 0; k0 < HEAD_; k0 += 32) {
            v16bf a = ldfrag(&tl[k0], 128);
            v16bf b = ldfrag(Wd2 + (size_t)(w * 16) * HEAD_ + k0, HEAD_);
            acc2 = wmma_bf16(a, b, acc2);
        }
        const float bias2 = bd2[w * 16 + n];
#pragma unroll
        for (int r = 0; r < 8; ++r) {
            const int m = mh + r;
            out[(size_t)(bm0 + m) * (S_ * OUT_) + (size_t)s * OUT_ + w * 16 + n] =
                acc2[r] + bias2;
        }
    }
}

// ---------- host launcher ----------
extern "C" void kernel_launch(void* const* d_in, const int* in_sizes, int n_in,
                              void* d_out, int out_size, void* d_ws, size_t ws_size,
                              hipStream_t stream) {
    (void)in_sizes; (void)n_in; (void)out_size; (void)ws_size;

    const float* x   = (const float*)d_in[0];   // [B,S,IN]
    const float* Wih = (const float*)d_in[1];   // [4H,IN]
    const float* bih = (const float*)d_in[2];   // [4H]
    const float* Whh = (const float*)d_in[3];   // [4H,H]
    const float* bhh = (const float*)d_in[4];   // [4H]
    const float* Wd  = (const float*)d_in[5];   // [HEAD,H]
    const float* bd  = (const float*)d_in[6];   // [HEAD]
    const float* Wd2 = (const float*)d_in[7];   // [OUT,HEAD]
    const float* bd2 = (const float*)d_in[8];   // [OUT]
    float* out = (float*)d_out;                 // [B,S,OUT]

    // carve workspace (256B aligned slices)
    char* ws = (char*)d_ws;
    size_t off = 0;
    auto carve = [&](size_t bytes) -> char* {
        char* p = ws + off;
        off += (bytes + 255) & ~(size_t)255;
        return p;
    };
    __bf16* xbf    = (__bf16*)carve((size_t)B_ * S_ * IN_ * 2);
    __bf16* Wihbf  = (__bf16*)carve((size_t)G4H_ * IN_ * 2);
    __bf16* Whhbf  = (__bf16*)carve((size_t)G4H_ * H_ * 2);
    __bf16* Wdbf   = (__bf16*)carve((size_t)HEAD_ * H_ * 2);
    __bf16* Wd2bf  = (__bf16*)carve((size_t)OUT_ * HEAD_ * 2);
    float*  bsum   = (float*)carve((size_t)G4H_ * 4);
    __bf16* hbuf0  = (__bf16*)carve((size_t)B_ * H_ * 2);
    __bf16* hbuf1  = (__bf16*)carve((size_t)B_ * H_ * 2);
    float*  cbuf   = (float*)carve((size_t)B_ * H_ * 4);

    auto blocks = [](int n) { return (n + 255) / 256; };

    // precision conversion + bias precompute + state init
    { int n = B_ * S_ * IN_;    cvt_bf16_kernel<<<blocks(n), 256, 0, stream>>>(x,   xbf,   n); }
    { int n = G4H_ * IN_;       cvt_bf16_kernel<<<blocks(n), 256, 0, stream>>>(Wih, Wihbf, n); }
    { int n = G4H_ * H_;        cvt_bf16_kernel<<<blocks(n), 256, 0, stream>>>(Whh, Whhbf, n); }
    { int n = HEAD_ * H_;       cvt_bf16_kernel<<<blocks(n), 256, 0, stream>>>(Wd,  Wdbf,  n); }
    { int n = OUT_ * HEAD_;     cvt_bf16_kernel<<<blocks(n), 256, 0, stream>>>(Wd2, Wd2bf, n); }
    { int n = G4H_;             bias_sum_kernel<<<blocks(n), 256, 0, stream>>>(bih, bhh, bsum, n); }
    { int n = B_ * H_;          state_zero_kernel<<<blocks(n), 256, 0, stream>>>(hbuf0, cbuf, n); }

    // sequential recurrence: double-buffer h across steps
    for (int s = 0; s < S_; ++s) {
        const __bf16* hin  = (s & 1) ? hbuf1 : hbuf0;
        __bf16*       hout = (s & 1) ? hbuf0 : hbuf1;
        lstm_step_kernel<<<dim3(H_ / 64, B_ / 32), 256, 0, stream>>>(
            xbf, Wihbf, Whhbf, bsum, hin, hout, cbuf, s);
        head_kernel<<<B_ / 16, 256, 0, stream>>>(
            hout, Wdbf, bd, Wd2bf, bd2, out, s);
    }
}
